// BasicBlock_11227044512048
// MI455X (gfx1250) — matrix operand order repaired and checked by reading, not compiled, not run on hardware
//
#include <hip/hip_runtime.h>
#include <math.h>

// ---------------------------------------------------------------------------
// EfficientViT BasicBlock for MI455X (gfx1250, wave32, WMMA).
// Layout strategy: NHWC f32 main residual stream (fits L2: 103MB < 192MB),
// bf16 staging buffers for all WMMA GEMM operands, fused epilogues.
// Workspace: Xa + Xb (f32, 103MB ea) + H16 (bf16 hidden, 103MB) +
//            A16/Y16 (bf16, 51MB, reused) + bf16 weights (~5MB) ~= 365MB.
// ---------------------------------------------------------------------------

#define BATCH 64
#define EDIM  512
#define RESO  28
#define NPXT  (BATCH * RESO * RESO)          // 50176 pixels
#define NELEM ((size_t)NPXT * EDIM)          // 25,690,112 elements

typedef __attribute__((ext_vector_type(16))) __bf16 v16bf_t;
typedef __attribute__((ext_vector_type(8)))  float  v8f_t;
typedef __attribute__((ext_vector_type(4)))  int    i4_t;

__device__ __forceinline__ unsigned short f2bf(float f) {
  unsigned u = __float_as_uint(f);
  u += 0x7fffu + ((u >> 16) & 1u);           // round-to-nearest-even
  return (unsigned short)(u >> 16);
}
__device__ __forceinline__ float bf2f(unsigned short h) {
  return __uint_as_float(((unsigned)h) << 16);
}
__device__ __forceinline__ float hswish(float x) {
  return x * fminf(fmaxf(x + 3.0f, 0.0f), 6.0f) * (1.0f / 6.0f);
}

// ------------------------------ layout converts ----------------------------
__global__ __launch_bounds__(256) void k_nchw2nhwc(const float* __restrict__ in,
                                                   float* __restrict__ out) {
  size_t idx = (size_t)blockIdx.x * 256 + threadIdx.x;
  int w = (int)(idx % RESO);
  size_t r1 = idx / RESO;
  int h = (int)(r1 % RESO);
  size_t r2 = r1 / RESO;
  int c = (int)(r2 % EDIM);
  int b = (int)(r2 / EDIM);
  out[(((size_t)b * RESO + h) * RESO + w) * EDIM + c] = in[idx];
}

__global__ __launch_bounds__(256) void k_nhwc2nchw(const float* __restrict__ in,
                                                   float* __restrict__ out) {
  size_t idx = (size_t)blockIdx.x * 256 + threadIdx.x;
  int c = (int)(idx % EDIM);
  size_t r1 = idx / EDIM;
  int w = (int)(r1 % RESO);
  size_t r2 = r1 / RESO;
  int h = (int)(r2 % RESO);
  int b = (int)(r2 / RESO);
  out[(((size_t)b * EDIM + c) * RESO + h) * RESO + w] = in[idx];
}

__global__ __launch_bounds__(256) void k_f2bf(const float* __restrict__ in,
                                              unsigned short* __restrict__ out, int n) {
  int i = blockIdx.x * 256 + threadIdx.x;
  if (i < n) out[i] = f2bf(in[i]);
}

// ---------------- fused 3x3 depthwise conv + bias + residual ---------------
// NHWC; one block per pixel, threads over channels. Emits f32 result and
// a bf16 copy (the A operand of the following WMMA GEMM).
__global__ __launch_bounds__(256) void k_dw3res(const float* __restrict__ X,
                                                const float* __restrict__ W,
                                                const float* __restrict__ Bv,
                                                float* __restrict__ Xo,
                                                unsigned short* __restrict__ A16) {
  int p = blockIdx.x;
  int b = p / (RESO * RESO);
  int r = p - b * (RESO * RESO);
  int h = r / RESO;
  int wc = r - h * RESO;
  size_t base = (size_t)p * EDIM;
  for (int c = threadIdx.x; c < EDIM; c += 256) {
    float acc = Bv[c];
#pragma unroll
    for (int dy = -1; dy <= 1; ++dy) {
      int hh = h + dy;
      if (hh < 0 || hh >= RESO) continue;
#pragma unroll
      for (int dx = -1; dx <= 1; ++dx) {
        int ww = wc + dx;
        if (ww < 0 || ww >= RESO) continue;
        size_t np = ((size_t)b * (RESO * RESO) + hh * RESO + ww) * EDIM;
        acc += X[np + c] * W[c * 9 + (dy + 1) * 3 + (dx + 1)];
      }
    }
    float o = X[base + c] + acc;
    Xo[base + c] = o;
    A16[base + c] = f2bf(o);
  }
}

// ------------------------------- WMMA GEMM ---------------------------------
// out[m,n] = sum_k A[m,k] * W[n,k]  (A: [M,K] bf16 row-major activations,
// W: [N,K] bf16 row-major weights — both K-contiguous, ideal for the
// 16x16x32 bf16 WMMA fragment layouts).
// Block: 256 thr = 8 waves; tile 64(M) x 128(N); wave tile 16 x 64.
// ep==1: out_bf16 = hswish(acc + bias[n])          (FFN first 1x1)
// ep==2: out_f32  = acc + bias[n] + res[m,n]       (FFN second 1x1 / proj)
__global__ __launch_bounds__(256) void k_gemm_bf16(
    const unsigned short* __restrict__ A, const unsigned short* __restrict__ W,
    const float* __restrict__ bias, const float* __restrict__ res,
    float* __restrict__ outF, unsigned short* __restrict__ outB,
    int M, int N, int K, int ep) {
  const int lane = threadIdx.x & 31;
  const int wave = threadIdx.x >> 5;
  const int wm = wave >> 1;                  // 0..3
  const int wn = wave & 1;                   // 0..1
  const int m0 = blockIdx.x * 64 + wm * 16;
  const int n0 = blockIdx.y * 128 + wn * 64;
  const int l15 = lane & 15;
  const int half = lane >> 4;                // 0 or 1

  // A 16x32 bf16 fragment: lanes 0-15 hold K=[k0..k0+15],
  // lanes 16-31 hold K=[k0+16..k0+31].
  const unsigned short* Ap = A + (size_t)(m0 + l15) * K + half * 16;
  // B 32x16 bf16 fragment: lane's column = n, 16 contiguous K per half-wave.
  const unsigned short* Wp0 = W + (size_t)(n0 + 0  + l15) * K + half * 16;
  const unsigned short* Wp1 = W + (size_t)(n0 + 16 + l15) * K + half * 16;
  const unsigned short* Wp2 = W + (size_t)(n0 + 32 + l15) * K + half * 16;
  const unsigned short* Wp3 = W + (size_t)(n0 + 48 + l15) * K + half * 16;

  v8f_t acc[4] = {};

  for (int k0 = 0; k0 < K; k0 += 32) {
    union { v16bf_t v; i4_t i[2]; } a, b0, b1, b2, b3;
    a.i[0]  = *(const i4_t*)(Ap  + k0);
    a.i[1]  = *(const i4_t*)(Ap  + k0 + 8);
    b0.i[0] = *(const i4_t*)(Wp0 + k0);
    b0.i[1] = *(const i4_t*)(Wp0 + k0 + 8);
    b1.i[0] = *(const i4_t*)(Wp1 + k0);
    b1.i[1] = *(const i4_t*)(Wp1 + k0 + 8);
    b2.i[0] = *(const i4_t*)(Wp2 + k0);
    b2.i[1] = *(const i4_t*)(Wp2 + k0 + 8);
    b3.i[0] = *(const i4_t*)(Wp3 + k0);
    b3.i[1] = *(const i4_t*)(Wp3 + k0 + 8);
    acc[0] = __builtin_amdgcn_wmma_f32_16x16x32_bf16(false, a.v, false, b0.v,
                                                     (short)0, acc[0], false, false);
    acc[1] = __builtin_amdgcn_wmma_f32_16x16x32_bf16(false, a.v, false, b1.v,
                                                     (short)0, acc[1], false, false);
    acc[2] = __builtin_amdgcn_wmma_f32_16x16x32_bf16(false, a.v, false, b2.v,
                                                     (short)0, acc[2], false, false);
    acc[3] = __builtin_amdgcn_wmma_f32_16x16x32_bf16(false, a.v, false, b3.v,
                                                     (short)0, acc[3], false, false);
  }

  // C/D layout: VGPR r -> m = m0 + r + 8*half, lane%16 -> n.
  const int mb = m0 + half * 8;
#pragma unroll
  for (int tt = 0; tt < 4; ++tt) {
    int n = n0 + tt * 16 + l15;
    float bn = bias[n];
#pragma unroll
    for (int r = 0; r < 8; ++r) {
      int m = mb + r;
      size_t off = (size_t)m * N + n;
      float v = acc[tt][r] + bn;
      if (ep == 1) {
        outB[off] = f2bf(hswish(v));
      } else {
        outF[off] = v + res[off];
      }
    }
  }
}

// --------------------------- cascade attention -----------------------------
// One 7x7 window per block; all 8 heads sequential in LDS (cascade).
// Writes Y16 = bf16(hswish(concat head outputs)) in NHWC pixel order, which
// is the A operand of the proj WMMA GEMM.
__global__ __launch_bounds__(256) void k_cascade_attn(
    const float* __restrict__ X, const float* __restrict__ qkv_w,
    const float* __restrict__ qkv_b,
    const float* __restrict__ w7, const float* __restrict__ b7,
    const float* __restrict__ w5, const float* __restrict__ b5,
    const float* __restrict__ w3, const float* __restrict__ b3,
    const float* __restrict__ abias, unsigned short* __restrict__ Y16) {
  __shared__ float sp[49 * 64];   // cascade carrier (f32 exact)
  __shared__ float vb[49 * 64];
  __shared__ float qb[49 * 16];
  __shared__ float kb[49 * 16];
  __shared__ float qc[49 * 16];
  __shared__ float att[49 * 49];
  __shared__ int   pix[49];

  const int t = threadIdx.x;
  const int wi = blockIdx.x;                 // 0..1023 windows
  const int b  = wi >> 4;
  const int wh = (wi >> 2) & 3;
  const int ww = wi & 3;

  if (t < 49) {
    int r = t / 7, c = t - (t / 7) * 7;
    pix[t] = (b * RESO + wh * 7 + r) * RESO + (ww * 7 + c);
  }
  __syncthreads();

  for (int head = 0; head < 8; ++head) {
    // (a) sp = (head==0 ? chunk0 : prev_attn_out + chunk_head)
    for (int e = t; e < 49 * 64; e += 256) {
      int p = e >> 6, c = e & 63;
      float xv = X[(size_t)pix[p] * EDIM + head * 64 + c];
      sp[e] = (head == 0) ? xv : sp[e] + xv;
    }
    __syncthreads();
    // (b) qkv 1x1: 96 outputs per pixel, K=64
    const float* Wq = qkv_w + head * 96 * 64;
    const float* Bq = qkv_b + head * 96;
    for (int e = t; e < 49 * 96; e += 256) {
      int p = e / 96, o = e - p * 96;
      const float* wr = Wq + o * 64;
      const float* sr = sp + p * 64;
      float acc = Bq[o];
      for (int c2 = 0; c2 < 64; ++c2) acc += sr[c2] * wr[c2];
      if (o < 16) qb[p * 16 + o] = acc;
      else if (o < 32) kb[p * 16 + (o - 16)] = acc;
      else vb[p * 64 + (o - 32)] = acc;
    }
    __syncthreads();
    // (c) depthwise conv on q within the 7x7 window, SAME padding
    int ks = (head == 0) ? 7 : (head == 1) ? 5 : 3;
    const float* cw = (head == 0) ? w7 : (head == 1) ? w5 : (w3 + (head - 2) * 16 * 9);
    const float* cb = (head == 0) ? b7 : (head == 1) ? b5 : (b3 + (head - 2) * 16);
    int pad = ks >> 1;
    for (int e = t; e < 49 * 16; e += 256) {
      int p = e >> 4, c = e & 15;
      int pr = p / 7, pc = p - (p / 7) * 7;
      float acc = cb[c];
      for (int kr = 0; kr < ks; ++kr) {
        int ir = pr + kr - pad;
        if (ir < 0 || ir >= 7) continue;
        for (int kc = 0; kc < ks; ++kc) {
          int ic = pc + kc - pad;
          if (ic < 0 || ic >= 7) continue;
          acc += qb[(ir * 7 + ic) * 16 + c] * cw[(c * ks + kr) * ks + kc];
        }
      }
      qc[e] = acc;
    }
    __syncthreads();
    // (d) logits: attn[n,m] = 0.25 * <qc[n], kb[m]> + bias[|dr|*7+|dc|]
    const float* ab = abias + head * 49;     // N_OFFS == 49, offs[(a,b)] = a*7+b
    for (int e = t; e < 49 * 49; e += 256) {
      int n = e / 49, m = e - n * 49;
      const float* qn = qc + n * 16;
      const float* km = kb + m * 16;
      float acc = 0.f;
      for (int c2 = 0; c2 < 16; ++c2) acc += qn[c2] * km[c2];
      int drr = n / 7 - m / 7; if (drr < 0) drr = -drr;
      int dcc = (n - (n / 7) * 7) - (m - (m / 7) * 7); if (dcc < 0) dcc = -dcc;
      att[e] = acc * 0.25f + ab[drr * 7 + dcc];
    }
    __syncthreads();
    // (e) softmax over m, thread-per-row
    if (t < 49) {
      float* row = att + t * 49;
      float mx = row[0];
      for (int m = 1; m < 49; ++m) mx = fmaxf(mx, row[m]);
      float s = 0.f;
      for (int m = 0; m < 49; ++m) { float ev = __expf(row[m] - mx); row[m] = ev; s += ev; }
      float inv = 1.0f / s;
      for (int m = 0; m < 49; ++m) row[m] *= inv;
    }
    __syncthreads();
    // (f) out[n,d] = sum_m attn[n,m] * v[m,d]; carrier for next head + Y16
    for (int e = t; e < 49 * 64; e += 256) {
      int p = e >> 6, d = e & 63;
      const float* ar = att + p * 49;
      float acc = 0.f;
      for (int m = 0; m < 49; ++m) acc += ar[m] * vb[m * 64 + d];
      sp[e] = acc;
      Y16[(size_t)pix[p] * EDIM + head * 64 + d] = f2bf(hswish(acc));
    }
    __syncthreads();
  }
}

// ------------------------------- launcher ----------------------------------
extern "C" void kernel_launch(void* const* d_in, const int* in_sizes, int n_in,
                              void* d_out, int out_size, void* d_ws, size_t ws_size,
                              hipStream_t stream) {
  (void)in_sizes; (void)n_in; (void)out_size; (void)ws_size;
  const float* x      = (const float*)d_in[0];
  const float* dw0_w  = (const float*)d_in[1];
  const float* dw0_b  = (const float*)d_in[2];
  const float* f0w1   = (const float*)d_in[3];
  const float* f0b1   = (const float*)d_in[4];
  const float* f0w2   = (const float*)d_in[5];
  const float* f0b2   = (const float*)d_in[6];
  const float* qkv_w  = (const float*)d_in[7];
  const float* qkv_b  = (const float*)d_in[8];
  const float* w7     = (const float*)d_in[9];
  const float* b7     = (const float*)d_in[10];
  const float* w5     = (const float*)d_in[11];
  const float* b5     = (const float*)d_in[12];
  const float* w3     = (const float*)d_in[13];
  const float* b3     = (const float*)d_in[14];
  const float* abias  = (const float*)d_in[15];
  const float* proj_w = (const float*)d_in[16];
  const float* proj_b = (const float*)d_in[17];
  const float* dw1_w  = (const float*)d_in[18];
  const float* dw1_b  = (const float*)d_in[19];
  const float* f1w1   = (const float*)d_in[20];
  const float* f1b1   = (const float*)d_in[21];
  const float* f1w2   = (const float*)d_in[22];
  const float* f1b2   = (const float*)d_in[23];
  float* out = (float*)d_out;

  char* ws = (char*)d_ws;
  size_t off = 0;
  auto take = [&](size_t bytes) -> void* {
    void* p = ws + off;
    off += (bytes + 255) & ~(size_t)255;
    return p;
  };
  float* Xa = (float*)take(NELEM * 4);
  float* Xb = (float*)take(NELEM * 4);
  unsigned short* H16 = (unsigned short*)take((size_t)NPXT * 1024 * 2);
  unsigned short* A16 = (unsigned short*)take(NELEM * 2);     // also reused as Y16
  unsigned short* w1a = (unsigned short*)take((size_t)1024 * 512 * 2);
  unsigned short* w2a = (unsigned short*)take((size_t)512 * 1024 * 2);
  unsigned short* w1b = (unsigned short*)take((size_t)1024 * 512 * 2);
  unsigned short* w2b = (unsigned short*)take((size_t)512 * 1024 * 2);
  unsigned short* pjw = (unsigned short*)take((size_t)512 * 512 * 2);

  dim3 blk(256);
  const int NCONV = (int)(NELEM / 256);                       // 100352, exact

  // weights -> bf16 (tiny, once per launch; deterministic)
  k_f2bf<<<dim3((1024 * 512 + 255) / 256), blk, 0, stream>>>(f0w1, w1a, 1024 * 512);
  k_f2bf<<<dim3((512 * 1024 + 255) / 256), blk, 0, stream>>>(f0w2, w2a, 512 * 1024);
  k_f2bf<<<dim3((1024 * 512 + 255) / 256), blk, 0, stream>>>(f1w1, w1b, 1024 * 512);
  k_f2bf<<<dim3((512 * 1024 + 255) / 256), blk, 0, stream>>>(f1w2, w2b, 512 * 1024);
  k_f2bf<<<dim3((512 * 512 + 255) / 256), blk, 0, stream>>>(proj_w, pjw, 512 * 512);

  // x (NCHW) -> Xa (NHWC)
  k_nchw2nhwc<<<dim3(NCONV), blk, 0, stream>>>(x, Xa);

  // x1 = x + dw0(x): Xa -> Xb (+ bf16 A16)
  k_dw3res<<<dim3(NPXT), blk, 0, stream>>>(Xa, dw0_w, dw0_b, Xb, A16);

  // ffn0: h = hswish(x1 W1^T + b1); x2 = x1 + h W2^T + b2  : -> Xa
  k_gemm_bf16<<<dim3(NPXT / 64, 1024 / 128), blk, 0, stream>>>(
      A16, w1a, f0b1, nullptr, nullptr, H16, NPXT, 1024, 512, 1);
  k_gemm_bf16<<<dim3(NPXT / 64, 512 / 128), blk, 0, stream>>>(
      H16, w2a, f0b2, Xb, Xa, nullptr, NPXT, 512, 1024, 2);

  // cascade attention over 7x7 windows: Xa -> Y16 (=A16, hswish'ed, bf16)
  k_cascade_attn<<<dim3(1024), blk, 0, stream>>>(
      Xa, qkv_w, qkv_b, w7, b7, w5, b5, w3, b3, abias, A16);

  // x3 = x2 + proj(y): -> Xb
  k_gemm_bf16<<<dim3(NPXT / 64, 512 / 128), blk, 0, stream>>>(
      A16, pjw, proj_b, Xa, Xb, nullptr, NPXT, 512, 512, 2);

  // x4 = x3 + dw1(x3): Xb -> Xa (+ bf16 A16)
  k_dw3res<<<dim3(NPXT), blk, 0, stream>>>(Xb, dw1_w, dw1_b, Xa, A16);

  // ffn1: -> Xb
  k_gemm_bf16<<<dim3(NPXT / 64, 1024 / 128), blk, 0, stream>>>(
      A16, w1b, f1b1, nullptr, nullptr, H16, NPXT, 1024, 512, 1);
  k_gemm_bf16<<<dim3(NPXT / 64, 512 / 128), blk, 0, stream>>>(
      H16, w2b, f1b2, Xa, Xb, nullptr, NPXT, 512, 1024, 2);

  // Xb (NHWC) -> out (NCHW)
  k_nhwc2nchw<<<dim3(NCONV), blk, 0, stream>>>(Xb, out);
}
